// MoELayer_12051678232653
// MI455X (gfx1250) — matrix operand order repaired and checked by reading, not compiled, not run on hardware
//
#include <hip/hip_runtime.h>
#include <hip/hip_bf16.h>
#include <math.h>

// ---------------- problem constants ----------------
#define T_TOK 8192      // B*S tokens
#define D_DIM 1024
#define E_EXP 8
#define H_DIM 2736
#define HP_DIM 2752     // H padded to multiple of 64 (43 * 64)

typedef __bf16 bf16_t;
typedef __attribute__((ext_vector_type(16))) __bf16 v16bf;
typedef __attribute__((ext_vector_type(8)))  __bf16 v8bf;
typedef __attribute__((ext_vector_type(8)))  float  v8f;
typedef __attribute__((ext_vector_type(4)))  float  v4f;

// ---------------- WMMA helpers ----------------
__device__ __forceinline__ v8f wmma_bf16(v16bf a, v16bf b, v8f c) {
  // D = A(16x32) * B(32x16) + C ; bf16 inputs, f32 accum
  return __builtin_amdgcn_wmma_f32_16x16x32_bf16(
      /*neg_a=*/false, a, /*neg_b=*/false, b,
      /*c_mod=*/(short)0, c, /*reuse_a=*/false, /*reuse_b=*/false);
}

// Pack two 8-half LDS reads into a 16-half fragment register set.
__device__ __forceinline__ v16bf load_frag16(const bf16_t* base, int off0, int off1) {
  v8bf lo = *(const v8bf*)(base + off0);
  v8bf hi = *(const v8bf*)(base + off1);
  v16bf r;
#pragma unroll
  for (int i = 0; i < 8; ++i) { r[i] = lo[i]; r[i + 8] = hi[i]; }
  return r;
}

// CDNA5 async global->LDS copy: 16 bytes per lane, bypasses VGPRs, ASYNCcnt.
__device__ __forceinline__ void async_ld16(const bf16_t* gptr, bf16_t* lptr) {
  unsigned lds_off = (unsigned)(uintptr_t)lptr;  // low 32 bits = LDS byte offset
  asm volatile("global_load_async_to_lds_b128 %0, %1, off"
               :: "v"(lds_off), "v"(gptr) : "memory");
}

#define WAIT_ASYNC(n) asm volatile("s_wait_asynccnt %0" :: "i"(n) : "memory")

// ---------------- init / convert kernels ----------------
__global__ void zero_out_kernel(float* __restrict__ out) {
  size_t i4 = ((size_t)blockIdx.x * blockDim.x + threadIdx.x) * 4;
  if (i4 + 3 < (size_t)T_TOK * D_DIM) {
    *(v4f*)(out + i4) = (v4f){0.f, 0.f, 0.f, 0.f};
  }
}

__global__ void cvt_x_kernel(const float* __restrict__ x, bf16_t* __restrict__ xb) {
  size_t i4 = ((size_t)blockIdx.x * blockDim.x + threadIdx.x) * 4;
  v4f v = *(const v4f*)(x + i4);
#pragma unroll
  for (int j = 0; j < 4; ++j) xb[i4 + j] = (bf16_t)v[j];
}

// Wg/W1 for one expert: [H, D] fp32 -> [HP, D] bf16 (zero pad rows H..HP-1)
__global__ void cvt_wpair_kernel(const float* __restrict__ wg, const float* __restrict__ w1,
                                 bf16_t* __restrict__ gb, bf16_t* __restrict__ ub) {
  size_t lin = ((size_t)blockIdx.x * blockDim.x + threadIdx.x) * 4;
  int h = (int)(lin >> 10);          // D == 1024
  int d = (int)(lin & 1023);
  if (h < H_DIM) {
    const float* pg = wg + (size_t)h * D_DIM + d;
    const float* pu = w1 + (size_t)h * D_DIM + d;
#pragma unroll
    for (int j = 0; j < 4; ++j) {
      gb[lin + j] = (bf16_t)pg[j];
      ub[lin + j] = (bf16_t)pu[j];
    }
  } else {
#pragma unroll
    for (int j = 0; j < 4; ++j) { gb[lin + j] = (bf16_t)0.f; ub[lin + j] = (bf16_t)0.f; }
  }
}

// W2 for one expert: [D, H] fp32 -> [D, HP] bf16 (zero pad cols H..HP-1)
__global__ void cvt_w2_kernel(const float* __restrict__ w2, bf16_t* __restrict__ w2b) {
  size_t lin = ((size_t)blockIdx.x * blockDim.x + threadIdx.x) * 4;
  int d = (int)(lin / HP_DIM);
  int k = (int)(lin % HP_DIM);
#pragma unroll
  for (int j = 0; j < 4; ++j) {
    int kk = k + j;
    w2b[lin + j] = (kk < H_DIM) ? (bf16_t)w2[(size_t)d * H_DIM + kk] : (bf16_t)0.f;
  }
}

// ---------------- router: softmax + top-2 combine weights ----------------
__global__ void router_kernel(const float* __restrict__ x, const float* __restrict__ gw,
                              float* __restrict__ combine) {
  const int lane = threadIdx.x & 31;
  const int wave = threadIdx.x >> 5;
  const int t = blockIdx.x * 8 + wave;          // 8 waves/block, one token per wave

  float acc[E_EXP];
#pragma unroll
  for (int e = 0; e < E_EXP; ++e) acc[e] = 0.f;

  const float* xt = x + (size_t)t * D_DIM;
  for (int d = lane; d < D_DIM; d += 32) {
    float xv = xt[d];
#pragma unroll
    for (int e = 0; e < E_EXP; ++e) acc[e] += xv * gw[e * D_DIM + d];
  }
  // wave32 tree reduction
#pragma unroll
  for (int e = 0; e < E_EXP; ++e) {
#pragma unroll
    for (int off = 16; off > 0; off >>= 1) acc[e] += __shfl_xor(acc[e], off, 32);
  }

  if (lane == 0) {
    float m = acc[0];
#pragma unroll
    for (int e = 1; e < E_EXP; ++e) m = fmaxf(m, acc[e]);
    float p[E_EXP], s = 0.f;
#pragma unroll
    for (int e = 0; e < E_EXP; ++e) { p[e] = __expf(acc[e] - m); s += p[e]; }
    float inv = 1.f / s;
#pragma unroll
    for (int e = 0; e < E_EXP; ++e) p[e] *= inv;
    // top-2, first index wins ties (matches jax.lax.top_k)
    int i1 = 0;
#pragma unroll
    for (int e = 1; e < E_EXP; ++e) if (p[e] > p[i1]) i1 = e;
    int i2 = (i1 == 0) ? 1 : 0;
#pragma unroll
    for (int e = 0; e < E_EXP; ++e) if (e != i1 && p[e] > p[i2]) i2 = e;
    float denom = p[i1] + p[i2] + 1e-8f;
    float w1 = p[i1] / denom, w2 = p[i2] / denom;
#pragma unroll
    for (int e = 0; e < E_EXP; ++e)
      combine[(size_t)t * E_EXP + e] = (e == i1) ? w1 : ((e == i2) ? w2 : 0.f);
  }
}

// ---------------- gemmA: A = silu(X*Wg^T) .* (X*W1^T), bf16 out ----------------
// block tile: 128(M) x 64(N); 8 waves as 4(M) x 2(N); wave tile 32x32; K-step 32
// Triple-buffered LDS fed by global_load_async_to_lds_b128; one barrier + one
// s_wait_asynccnt per K-step; last two steps peeled so the hot loop is branch-free.
#define LDSP 40   // padded row length in halves (16B-aligned fragment reads)

__global__ __launch_bounds__(256) void gemmA_kernel(const bf16_t* __restrict__ Xb,
                                                    const bf16_t* __restrict__ Gw,
                                                    const bf16_t* __restrict__ Uw,
                                                    bf16_t* __restrict__ Abuf) {
  __shared__ bf16_t Xs[3][128][LDSP];
  __shared__ bf16_t Gs[3][64][LDSP];
  __shared__ bf16_t Us[3][64][LDSP];

  const int tid  = threadIdx.x;
  const int lane = tid & 31;
  const int wave = tid >> 5;
  const int wm = wave & 3;        // 0..3 -> M quadrant
  const int wn = wave >> 2;       // 0..1 -> N half
  const int rowBase = blockIdx.x * 128;
  const int colBase = blockIdx.y * 64;

  v8f accg[2][2], accu[2][2];
#pragma unroll
  for (int i = 0; i < 2; ++i)
#pragma unroll
    for (int j = 0; j < 2; ++j) {
      accg[i][j] = (v8f)(0.f);
      accu[i][j] = (v8f)(0.f);
    }

  const int kh  = (lane >> 4) * 8;    // unused; A and B frags both use klo
  const int klo = (lane >> 4) * 16;   // B-frag K sub-base
  const int am  = lane & 15;

  // per K-step staging: X 128x32 halves (2 segs/thread) + 2 weight tiles (1 each)
  // => 4 async instructions per wave per stage
  const int xr0 = tid >> 2,          xc0 = (tid & 3) * 8;
  const int xr1 = (tid + 256) >> 2,  xc1 = ((tid + 256) & 3) * 8;
  const int wr  = tid >> 2,          wc  = (tid & 3) * 8;

  auto stage = [&](int buf, int k0) {
    async_ld16(&Xb[(size_t)(rowBase + xr0) * D_DIM + k0 + xc0], &Xs[buf][xr0][xc0]);
    async_ld16(&Xb[(size_t)(rowBase + xr1) * D_DIM + k0 + xc1], &Xs[buf][xr1][xc1]);
    async_ld16(&Gw[(size_t)(colBase + wr) * D_DIM + k0 + wc],   &Gs[buf][wr][wc]);
    async_ld16(&Uw[(size_t)(colBase + wr) * D_DIM + k0 + wc],   &Us[buf][wr][wc]);
  };

  auto compute = [&](int buf) {
    const int bn = wn * 32 + am;
    v16bf a0 = load_frag16(&Xs[buf][wm * 32 + am][0],      klo, klo + 8);
    v16bf a1 = load_frag16(&Xs[buf][wm * 32 + 16 + am][0], klo, klo + 8);
    v16bf bg0 = load_frag16(&Gs[buf][bn][0],      klo, klo + 8);
    v16bf bg1 = load_frag16(&Gs[buf][bn + 16][0], klo, klo + 8);
    v16bf bu0 = load_frag16(&Us[buf][bn][0],      klo, klo + 8);
    v16bf bu1 = load_frag16(&Us[buf][bn + 16][0], klo, klo + 8);
    accg[0][0] = wmma_bf16(a0, bg0, accg[0][0]);
    accg[0][1] = wmma_bf16(a0, bg1, accg[0][1]);
    accg[1][0] = wmma_bf16(a1, bg0, accg[1][0]);
    accg[1][1] = wmma_bf16(a1, bg1, accg[1][1]);
    accu[0][0] = wmma_bf16(a0, bu0, accu[0][0]);
    accu[0][1] = wmma_bf16(a0, bu1, accu[0][1]);
    accu[1][0] = wmma_bf16(a1, bu0, accu[1][0]);
    accu[1][1] = wmma_bf16(a1, bu1, accu[1][1]);
  };

  const int NK = D_DIM / 32;          // 32 K-steps
  stage(0, 0);
  stage(1, 32);
  int bc = 0, bs = 2, kstage = 64;
  for (int i = 0; i < NK - 2; ++i) {
    WAIT_ASYNC(4);                    // drain stage i (stage i+1 stays in flight)
    __syncthreads();                  // everyone's stage i landed; buf (i-1)%3 free
    stage(bs, kstage);
    kstage += 32;
    compute(bc);
    bc = (bc == 2) ? 0 : bc + 1;
    bs = (bs == 2) ? 0 : bs + 1;
  }
  WAIT_ASYNC(4);
  __syncthreads();
  compute(bc);
  bc = (bc == 2) ? 0 : bc + 1;
  WAIT_ASYNC(0);
  __syncthreads();
  compute(bc);

  // epilogue: silu(g) * up  -> bf16
#pragma unroll
  for (int mt = 0; mt < 2; ++mt)
#pragma unroll
    for (int nt = 0; nt < 2; ++nt) {
      int col = colBase + wn * 32 + nt * 16 + am;
#pragma unroll
      for (int v = 0; v < 8; ++v) {
        int row = rowBase + wm * 32 + mt * 16 + v + 8 * (lane >> 4);
        float g = accg[mt][nt][v];
        float u = accu[mt][nt][v];
        float y = (g / (1.f + __expf(-g))) * u;
        Abuf[(size_t)row * HP_DIM + col] = (bf16_t)y;
      }
    }
}

// ---------------- gemmB: out += combine[:,e] .* (A @ W2^T) ----------------
__global__ __launch_bounds__(256) void gemmB_kernel(const bf16_t* __restrict__ Abuf,
                                                    const bf16_t* __restrict__ W2b,
                                                    const float* __restrict__ combine,
                                                    float* __restrict__ out, int expert) {
  __shared__ bf16_t As[3][128][LDSP];
  __shared__ bf16_t Ws[3][64][LDSP];

  const int tid  = threadIdx.x;
  const int lane = tid & 31;
  const int wave = tid >> 5;
  const int wm = wave & 3;
  const int wn = wave >> 2;
  const int rowBase = blockIdx.x * 128;
  const int colBase = blockIdx.y * 64;

  v8f acc[2][2];
#pragma unroll
  for (int i = 0; i < 2; ++i)
#pragma unroll
    for (int j = 0; j < 2; ++j) acc[i][j] = (v8f)(0.f);

  const int kh  = (lane >> 4) * 8;
  const int klo = (lane >> 4) * 16;
  const int am  = lane & 15;

  const int xr0 = tid >> 2,          xc0 = (tid & 3) * 8;
  const int xr1 = (tid + 256) >> 2,  xc1 = ((tid + 256) & 3) * 8;
  const int wr  = tid >> 2,          wc  = (tid & 3) * 8;

  auto stage = [&](int buf, int k0) {
    async_ld16(&Abuf[(size_t)(rowBase + xr0) * HP_DIM + k0 + xc0], &As[buf][xr0][xc0]);
    async_ld16(&Abuf[(size_t)(rowBase + xr1) * HP_DIM + k0 + xc1], &As[buf][xr1][xc1]);
    async_ld16(&W2b[(size_t)(colBase + wr) * HP_DIM + k0 + wc],    &Ws[buf][wr][wc]);
  };

  auto compute = [&](int buf) {
    const int bn = wn * 32 + am;
    v16bf a0 = load_frag16(&As[buf][wm * 32 + am][0],      klo, klo + 8);
    v16bf a1 = load_frag16(&As[buf][wm * 32 + 16 + am][0], klo, klo + 8);
    v16bf b0 = load_frag16(&Ws[buf][bn][0],      klo, klo + 8);
    v16bf b1 = load_frag16(&Ws[buf][bn + 16][0], klo, klo + 8);
    acc[0][0] = wmma_bf16(a0, b0, acc[0][0]);
    acc[0][1] = wmma_bf16(a0, b1, acc[0][1]);
    acc[1][0] = wmma_bf16(a1, b0, acc[1][0]);
    acc[1][1] = wmma_bf16(a1, b1, acc[1][1]);
  };

  const int NK = HP_DIM / 32;        // 86 K-steps
  stage(0, 0);
  stage(1, 32);
  int bc = 0, bs = 2, kstage = 64;
  for (int i = 0; i < NK - 2; ++i) {
    WAIT_ASYNC(3);
    __syncthreads();
    stage(bs, kstage);
    kstage += 32;
    compute(bc);
    bc = (bc == 2) ? 0 : bc + 1;
    bs = (bs == 2) ? 0 : bs + 1;
  }
  WAIT_ASYNC(3);
  __syncthreads();
  compute(bc);
  bc = (bc == 2) ? 0 : bc + 1;
  WAIT_ASYNC(0);
  __syncthreads();
  compute(bc);

#pragma unroll
  for (int mt = 0; mt < 2; ++mt)
#pragma unroll
    for (int nt = 0; nt < 2; ++nt) {
      int col = colBase + wn * 32 + nt * 16 + am;
#pragma unroll
      for (int v = 0; v < 8; ++v) {
        int row = rowBase + wm * 32 + mt * 16 + v + 8 * (lane >> 4);
        float w = combine[(size_t)row * E_EXP + expert];
        out[(size_t)row * D_DIM + col] += w * acc[mt][nt][v];
      }
    }
}

// ---------------- host launch ----------------
extern "C" void kernel_launch(void* const* d_in, const int* in_sizes, int n_in,
                              void* d_out, int out_size, void* d_ws, size_t ws_size,
                              hipStream_t stream) {
  (void)in_sizes; (void)n_in; (void)out_size; (void)ws_size;
  const float* x      = (const float*)d_in[0];   // [B,S,D]
  const float* gate_w = (const float*)d_in[1];   // [E,D]
  const float* Wg     = (const float*)d_in[2];   // [E,H,D]
  const float* W1     = (const float*)d_in[3];   // [E,H,D]
  const float* W2     = (const float*)d_in[4];   // [E,D,H]
  float* out = (float*)d_out;                    // [B,S,D]

  // workspace carve-out (256B aligned slices)
  char* p = (char*)d_ws;
  auto carve = [&](size_t bytes) -> char* {
    char* r = p;
    p += (bytes + 255) & ~(size_t)255;
    return r;
  };
  float*  combine = (float*) carve((size_t)T_TOK * E_EXP * sizeof(float));
  bf16_t* Xb      = (bf16_t*)carve((size_t)T_TOK * D_DIM * sizeof(bf16_t));
  bf16_t* Gb      = (bf16_t*)carve((size_t)HP_DIM * D_DIM * sizeof(bf16_t));
  bf16_t* Ub      = (bf16_t*)carve((size_t)HP_DIM * D_DIM * sizeof(bf16_t));
  bf16_t* W2b     = (bf16_t*)carve((size_t)D_DIM * HP_DIM * sizeof(bf16_t));
  bf16_t* Abuf    = (bf16_t*)carve((size_t)T_TOK * HP_DIM * sizeof(bf16_t));

  const int threads = 256;

  zero_out_kernel<<<(T_TOK * D_DIM / 4 + threads - 1) / threads, threads, 0, stream>>>(out);
  router_kernel<<<T_TOK / 8, threads, 0, stream>>>(x, gate_w, combine);
  cvt_x_kernel<<<(T_TOK * D_DIM / 4) / threads, threads, 0, stream>>>(x, Xb);

  for (int e = 0; e < E_EXP; ++e) {
    cvt_wpair_kernel<<<((size_t)HP_DIM * D_DIM / 4) / threads, threads, 0, stream>>>(
        Wg + (size_t)e * H_DIM * D_DIM, W1 + (size_t)e * H_DIM * D_DIM, Gb, Ub);
    cvt_w2_kernel<<<((size_t)D_DIM * HP_DIM / 4) / threads, threads, 0, stream>>>(
        W2 + (size_t)e * D_DIM * H_DIM, W2b);
    gemmA_kernel<<<dim3(T_TOK / 128, HP_DIM / 64), threads, 0, stream>>>(Xb, Gb, Ub, Abuf);
    gemmB_kernel<<<dim3(T_TOK / 128, D_DIM / 64), threads, 0, stream>>>(
        Abuf, W2b, combine, out, e);
  }
}
